// Decoder_51101520888269
// MI455X (gfx1250) — matrix operand order repaired and checked by reading, not compiled, not run on hardware
//
#include <hip/hip_runtime.h>

typedef __attribute__((ext_vector_type(16))) __bf16 v16bf;
typedef __attribute__((ext_vector_type(8)))  __bf16 v8bf;
typedef __attribute__((ext_vector_type(8)))  float  v8f;

// Problem dims (fixed by reference)
#define TT 64
#define BB 16
#define SS 128
#define HH 512
#define EE 512
#define VV 32000

// GEMM block tiling
#define MBLK 128   // 8 waves x 16 rows
#define NBLK 128   // per-wave 8 n-tiles of 16

// ---------------------------------------------------------------------------
// WMMA tile loaders, wave32; A and B share one contiguous k order (not the ISA table order)
// A (16x32 bf16): lane L -> row (L&15); 16 contiguous K at (L>>4)*16
// B (32x16 bf16): lane L -> col (L&15); 16 contiguous K at (L>>4)*16
// C/D (16x16 f32): VGPR r -> row r + 8*(L>>4), col (L&15)
// ---------------------------------------------------------------------------
__device__ __forceinline__ v8f zero8() {
  v8f z;
#pragma unroll
  for (int i = 0; i < 8; ++i) z[i] = 0.0f;
  return z;
}

__device__ __forceinline__ v16bf load_a_bf(const __bf16* A, int ldk, int m0,
                                           int k0, int lane) {
  const __bf16* p = A + (size_t)(m0 + (lane & 15)) * ldk + k0 + ((lane >> 4) << 4);
  v16bf r;
  *(v8bf*)&r        = *(const v8bf*)p;        // K = base .. base+7
  *((v8bf*)&r + 1)  = *(const v8bf*)(p + 8);  // K = base+8 .. base+15
  return r;
}

__device__ __forceinline__ v16bf load_b_bf(const __bf16* W, int ldk, int n0,
                                           int k0, int lane) {
  // W is row-major (N x K); B[k][n] = W[n][k] -> 32B contiguous per lane
  const __bf16* p = W + (size_t)(n0 + (lane & 15)) * ldk + k0 + ((lane >> 4) << 4);
  return *(const v16bf*)p;
}

// ---------------------------------------------------------------------------
// C(MxN) = A(MxK) * W(NxK)^T + bias, bf16 in, f32 out.
// Block = 8 waves computes a 128x128 tile; B k-slab (128 cols x 32 k, 8KB)
// double-buffered in LDS and shared by all 8 waves; each wave owns a
// 16x128 strip (8 WMMA accumulators). Requires M%128==0, N%128==0, K%32==0.
// ---------------------------------------------------------------------------
__global__ __launch_bounds__(256) void gemm_bf16_lds(
    const __bf16* __restrict__ A, const __bf16* __restrict__ W,
    const float* __restrict__ bias, float* __restrict__ C,
    int M, int N, int K) {
  __shared__ alignas(32) __bf16 bsh[2][NBLK * 32];

  const int tid  = threadIdx.x;
  const int lane = tid & 31;
  const int wave = tid >> 5;                  // 0..7
  const int nblocks = N / NBLK;
  const int m0 = (blockIdx.x / nblocks) * MBLK + wave * 16;
  const int n0 = (blockIdx.x % nblocks) * NBLK;
  (void)M;

  v8f acc[8];
#pragma unroll
  for (int j = 0; j < 8; ++j) acc[j] = zero8();

  // cooperative B staging: thread -> (col, 16-elem segment) of the 128x32 slab
  const int c   = tid >> 1;                   // 0..127 (local col)
  const int seg = (tid & 1) << 4;             // 0 or 16 (k offset)
  const __bf16* wsrc = W + (size_t)(n0 + c) * K + seg;

  // preload k-chunk 0
  *(v16bf*)&bsh[0][c * 32 + seg] = *(const v16bf*)wsrc;

  const int nk = K >> 5;
  for (int kc = 0; kc < nk; ++kc) {
    __syncthreads();                          // staged slab visible to all
    if (kc + 1 < nk) {                        // stage next slab into other buffer
      *(v16bf*)&bsh[(kc + 1) & 1][c * 32 + seg] =
          *(const v16bf*)(wsrc + (size_t)(kc + 1) * 32);
    }
    const __bf16* bbuf = bsh[kc & 1];
    v16bf a = load_a_bf(A, K, m0, kc * 32, lane);
#pragma unroll
    for (int j = 0; j < 8; ++j) {
      const __bf16* p = bbuf + (j * 16 + (lane & 15)) * 32 + ((lane >> 4) << 4);
      v16bf b = *(const v16bf*)p;             // ds_load, 32B contiguous
      acc[j] = __builtin_amdgcn_wmma_f32_16x16x32_bf16(
          false, a, false, b, (short)0, acc[j], false, false);
    }
  }

  const int nc = lane & 15;
  const int mb = m0 + ((lane >> 4) << 3);
#pragma unroll
  for (int j = 0; j < 8; ++j) {
    const int col = n0 + j * 16 + nc;
    const float bj = bias[col];
#pragma unroll
    for (int r = 0; r < 8; ++r)
      C[(size_t)(mb + r) * N + col] = acc[j][r] + bj;
  }
}

// ---------------------------------------------------------------------------
// f32 -> bf16 conversion
// ---------------------------------------------------------------------------
__global__ void f32_to_bf16(const float* __restrict__ s, __bf16* __restrict__ d,
                            int n) {
  int i = blockIdx.x * blockDim.x + threadIdx.x;
  if (i < n) d[i] = (__bf16)s[i];
}

// Embedding gather: x_bf[row] = bf16(emb[tgt[row]]); row = t*B + b
__global__ void gather_emb(const int* __restrict__ tgt,
                           const float* __restrict__ emb,
                           __bf16* __restrict__ x_bf) {
  const int row = blockIdx.x;
  const int v = tgt[row];
  const float* src = emb + (size_t)v * EE;
  for (int e = threadIdx.x; e < EE; e += blockDim.x)
    x_bf[(size_t)row * EE + e] = (__bf16)src[e];
}

// ---------------------------------------------------------------------------
// GRU: single workgroup (16 waves), 64 sequential steps.
// gh = h @ w_hh^T via WMMA; wave w owns j-tiles {2w, 2w+1} and their
// (r,z,n) gate tiles at n = jt*16, 512+jt*16, 1024+jt*16 -> gates computed
// directly from accumulators, no gh staging.
// ---------------------------------------------------------------------------
__global__ __launch_bounds__(512) void gru_kernel(
    const float* __restrict__ gx,      // (T*B, 3H) = x@w_ih^T + b_ih
    const float* __restrict__ state,   // (B, H)
    const __bf16* __restrict__ whh,    // (3H, H) bf16
    const float* __restrict__ b_hh,    // (3H)
    __bf16* __restrict__ ss_bf,        // (T*B, H) out
    float* __restrict__ hT_out)        // (B, H) out (tail of d_out)
{
  __shared__ alignas(32) float  h_f[BB * HH];
  __shared__ alignas(32) __bf16 h_bf[BB * HH];

  const int tid  = threadIdx.x;
  const int lane = tid & 31;
  const int wave = tid >> 5;       // 0..15

  for (int i = tid; i < BB * HH; i += 512) {
    float v = state[i];
    h_f[i] = v;
    h_bf[i] = (__bf16)v;
  }
  __syncthreads();

  const int nc = lane & 15;
  const int mb = (lane >> 4) << 3;

  for (int t = 0; t < TT; ++t) {
    v8f acc[2][3];
#pragma unroll
    for (int u = 0; u < 2; ++u)
#pragma unroll
      for (int g = 0; g < 3; ++g) acc[u][g] = zero8();

    for (int k0 = 0; k0 < HH; k0 += 32) {
      v16bf a = load_a_bf(h_bf, HH, 0, k0, lane);
#pragma unroll
      for (int u = 0; u < 2; ++u) {
        const int jt = wave * 2 + u;
#pragma unroll
        for (int g = 0; g < 3; ++g) {
          v16bf b = load_b_bf(whh, HH, g * HH + jt * 16, k0, lane);
          acc[u][g] = __builtin_amdgcn_wmma_f32_16x16x32_bf16(
              false, a, false, b, (short)0, acc[u][g], false, false);
        }
      }
    }
    __syncthreads();   // all waves finished reading h_bf

#pragma unroll
    for (int u = 0; u < 2; ++u) {
      const int jt = wave * 2 + u;
      const int j  = jt * 16 + nc;
      const float bhr = b_hh[j];
      const float bhz = b_hh[HH + j];
      const float bhn = b_hh[2 * HH + j];
#pragma unroll
      for (int r8 = 0; r8 < 8; ++r8) {
        const int b = mb + r8;
        const float* gxr = gx + (size_t)(t * BB + b) * (3 * HH);
        const float hr = acc[u][0][r8] + bhr;
        const float hz = acc[u][1][r8] + bhz;
        const float hn = acc[u][2][r8] + bhn;
        const float rg = 1.0f / (1.0f + __expf(-(gxr[j] + hr)));
        const float zg = 1.0f / (1.0f + __expf(-(gxr[HH + j] + hz)));
        const float ng = tanhf(gxr[2 * HH + j] + rg * hn);
        const float h2 = (1.0f - zg) * ng + zg * h_f[b * HH + j];
        h_f[b * HH + j]  = h2;
        h_bf[b * HH + j] = (__bf16)h2;
        ss_bf[(size_t)(t * BB + b) * HH + j] = (__bf16)h2;
      }
    }
    __syncthreads();   // h updated before next step's WMMA reads
  }

  for (int i = tid; i < BB * HH; i += 512) hT_out[i] = h_f[i];
}

// ---------------------------------------------------------------------------
// Fused attention per (b,t): scores -> mask -> softmax -> context.
// scores[s] = v . tanh(wh[b,s,:] + wss[t,b,:]) + v_b ; mask s >= src_len[b]
// content[d] = sum_s attn[s] * outputs[s,b,d]
// ---------------------------------------------------------------------------
__global__ __launch_bounds__(512) void attn_kernel(
    const float* __restrict__ wh,       // (S*B, H) row = s*B+b
    const float* __restrict__ wss,      // (T*B, H) row = t*B+b
    const float* __restrict__ outputs,  // (S, B, 2H) f32
    const int* __restrict__ src_len,
    const float* __restrict__ v_w, const float* __restrict__ v_b,
    __bf16* __restrict__ content_bf)    // (T*B, 2H)
{
  __shared__ float wss_s[HH];
  __shared__ float v_s[HH];
  __shared__ float sc[SS];

  const int bt = blockIdx.x;          // = t*B + b
  const int b  = bt & (BB - 1);
  const int tid = threadIdx.x;
  const int lane = tid & 31;
  const int wave = tid >> 5;          // 0..15
  const int slen = src_len[b];
  const float vb = v_b[0];

  wss_s[tid] = wss[(size_t)bt * HH + tid];
  v_s[tid]   = v_w[tid];
  __syncthreads();

  // scores: each wave handles 8 source positions
#pragma unroll
  for (int i = 0; i < 8; ++i) {
    const int s = wave + 16 * i;
    const float* whr = wh + (size_t)(s * BB + b) * HH;
    float acc = 0.0f;
    for (int h = lane; h < HH; h += 32)
      acc += v_s[h] * tanhf(whr[h] + wss_s[h]);
#pragma unroll
    for (int m = 16; m; m >>= 1) acc += __shfl_xor(acc, m, 32);
    if (lane == 0) {
      float e = acc + vb;
      if (s >= slen) e = -1e9f;
      sc[s] = e;
    }
  }
  __syncthreads();

  // softmax over S=128 (wave 0)
  if (wave == 0) {
    float x0 = sc[lane], x1 = sc[lane + 32], x2 = sc[lane + 64], x3 = sc[lane + 96];
    float mx = fmaxf(fmaxf(x0, x1), fmaxf(x2, x3));
#pragma unroll
    for (int m = 16; m; m >>= 1) mx = fmaxf(mx, __shfl_xor(mx, m, 32));
    float e0 = __expf(x0 - mx), e1 = __expf(x1 - mx);
    float e2 = __expf(x2 - mx), e3 = __expf(x3 - mx);
    float sm = e0 + e1 + e2 + e3;
#pragma unroll
    for (int m = 16; m; m >>= 1) sm += __shfl_xor(sm, m, 32);
    const float inv = 1.0f / sm;
    sc[lane] = e0 * inv; sc[lane + 32] = e1 * inv;
    sc[lane + 64] = e2 * inv; sc[lane + 96] = e3 * inv;
  }
  __syncthreads();

  // context: (2H) weighted sum over S
  for (int d = tid; d < 2 * HH; d += 512) {
    const float* ob = outputs + (size_t)b * (2 * HH) + d;
    float acc = 0.0f;
    for (int s = 0; s < SS; ++s)
      acc += sc[s] * ob[(size_t)s * BB * 2 * HH];
    content_bf[(size_t)bt * (2 * HH) + d] = (__bf16)acc;
  }
}

// r = tanh(t1 + t2) -> bf16
__global__ void tanh_combine(const float* __restrict__ t1,
                             const float* __restrict__ t2,
                             __bf16* __restrict__ r, int n) {
  int i = blockIdx.x * blockDim.x + threadIdx.x;
  if (i < n) r[i] = (__bf16)tanhf(t1[i] + t2[i]);
}

// ---------------------------------------------------------------------------
extern "C" void kernel_launch(void* const* d_in, const int* in_sizes, int n_in,
                              void* d_out, int out_size, void* d_ws,
                              size_t ws_size, hipStream_t stream) {
  (void)in_sizes; (void)n_in; (void)out_size; (void)ws_size;

  const int*   tgt    = (const int*)  d_in[0];
  const float* state  = (const float*)d_in[1];
  const float* outs   = (const float*)d_in[2];
  const int*   srclen = (const int*)  d_in[3];
  /* d_in[4] teacher_forcing == 1 (unused) */
  const float* emb    = (const float*)d_in[5];
  const float* w_ih   = (const float*)d_in[6];
  const float* w_hh   = (const float*)d_in[7];
  const float* b_ih   = (const float*)d_in[8];
  const float* b_hh   = (const float*)d_in[9];
  const float* fc1_w  = (const float*)d_in[10];
  const float* fc1_b  = (const float*)d_in[11];
  const float* fc2_w  = (const float*)d_in[12];
  const float* fc2_b  = (const float*)d_in[13];
  const float* v_w    = (const float*)d_in[14];
  const float* v_b    = (const float*)d_in[15];
  const float* fc_w   = (const float*)d_in[16];
  const float* fc_b   = (const float*)d_in[17];
  const float* fh_w   = (const float*)d_in[18];
  const float* fh_b   = (const float*)d_in[19];
  const float* ws_w   = (const float*)d_in[20];
  const float* ws_b   = (const float*)d_in[21];
  float* out = (float*)d_out;

  // workspace carve-up (all chunk sizes are multiples of 256B)
  char* wp = (char*)d_ws;
  auto take = [&](size_t bytes) {
    char* p = wp; wp += (bytes + 255) & ~(size_t)255; return p;
  };
  __bf16* x_bf    = (__bf16*)take((size_t)TT * BB * EE * 2);
  __bf16* wih_bf  = (__bf16*)take((size_t)3 * HH * EE * 2);
  __bf16* whh_bf  = (__bf16*)take((size_t)3 * HH * HH * 2);
  __bf16* fc1_bf  = (__bf16*)take((size_t)HH * 2 * HH * 2);
  __bf16* fc2_bf  = (__bf16*)take((size_t)HH * HH * 2);
  __bf16* fcw_bf  = (__bf16*)take((size_t)HH * 2 * HH * 2);
  __bf16* fhw_bf  = (__bf16*)take((size_t)HH * HH * 2);
  __bf16* wsw_bf  = (__bf16*)take((size_t)VV * HH * 2);
  __bf16* outs_bf = (__bf16*)take((size_t)SS * BB * 2 * HH * 2);
  float*  gx_f    = (float*) take((size_t)TT * BB * 3 * HH * 4);
  __bf16* ss_bf   = (__bf16*)take((size_t)TT * BB * HH * 2);
  float*  wh_f    = (float*) take((size_t)SS * BB * HH * 4);
  float*  wss_f   = (float*) take((size_t)TT * BB * HH * 4);
  __bf16* cont_bf = (__bf16*)take((size_t)TT * BB * 2 * HH * 2);
  float*  tmp1    = (float*) take((size_t)TT * BB * HH * 4);
  float*  tmp2    = (float*) take((size_t)TT * BB * HH * 4);
  __bf16* r_bf    = (__bf16*)take((size_t)TT * BB * HH * 2);

  auto cvt = [&](const float* s, __bf16* d, int n) {
    f32_to_bf16<<<(n + 255) / 256, 256, 0, stream>>>(s, d, n);
  };
  auto gemm = [&](const __bf16* A, const __bf16* W, const float* bias,
                  float* C, int M, int N, int K) {
    int blocks = (M / MBLK) * (N / NBLK);
    gemm_bf16_lds<<<blocks, 256, 0, stream>>>(A, W, bias, C, M, N, K);
  };

  // 1) pack weights & activations to bf16
  cvt(w_ih, wih_bf, 3 * HH * EE);
  cvt(w_hh, whh_bf, 3 * HH * HH);
  cvt(fc1_w, fc1_bf, HH * 2 * HH);
  cvt(fc2_w, fc2_bf, HH * HH);
  cvt(fc_w, fcw_bf, HH * 2 * HH);
  cvt(fh_w, fhw_bf, HH * HH);
  cvt(ws_w, wsw_bf, VV * HH);
  cvt(outs, outs_bf, SS * BB * 2 * HH);
  gather_emb<<<TT * BB, 256, 0, stream>>>(tgt, emb, x_bf);

  // 2) gx = x @ w_ih^T + b_ih   (1024 x 1536 x 512)
  gemm(x_bf, wih_bf, b_ih, gx_f, TT * BB, 3 * HH, EE);
  // 3) wh = outputs @ fc1_w^T + fc1_b   (2048 x 512 x 1024)
  gemm(outs_bf, fc1_bf, fc1_b, wh_f, SS * BB, HH, 2 * HH);
  // 4) GRU recurrence (writes ss_bf and hT into d_out tail)
  gru_kernel<<<1, 512, 0, stream>>>(gx_f, state, whh_bf, b_hh, ss_bf,
                                    out + (size_t)TT * BB * VV);
  // 5) wss = ss @ fc2_w^T + fc2_b   (1024 x 512 x 512)
  gemm(ss_bf, fc2_bf, fc2_b, wss_f, TT * BB, HH, HH);
  // 6) fused attention -> content (bf16)
  attn_kernel<<<TT * BB, 512, 0, stream>>>(wh_f, wss_f, outs, srclen,
                                           v_w, v_b, cont_bf);
  // 7) r = tanh(content@fc_w^T + fc_b + ss@fh_w^T + fh_b)
  gemm(cont_bf, fcw_bf, fc_b, tmp1, TT * BB, HH, 2 * HH);
  gemm(ss_bf, fhw_bf, fh_b, tmp2, TT * BB, HH, HH);
  tanh_combine<<<(TT * BB * HH + 255) / 256, 256, 0, stream>>>(
      tmp1, tmp2, r_bf, TT * BB * HH);
  // 8) logits = r @ ws_w^T + ws_b   (1024 x 32000 x 512) -> d_out
  gemm(r_bf, wsw_bf, ws_b, out, TT * BB, VV, HH);
}
